// NSTM_20426864460396
// MI455X (gfx1250) — compile-verified, matrix operand order checked by reading, not run on hardware
//
#include <hip/hip_runtime.h>
#include <hip/hip_bf16.h>
#include <cstdint>

// ---------------------------------------------------------------------------
// NSTM forward for MI455X (gfx1250, wave32, WMMA).
// Big GEMMs use V_WMMA_F32_16X16X32_BF16 (bf16 operands keep fp32 exponent
// range -- kernel entries go down to e^-40 which underflows fp16 -- with fp32
// accumulation). lax.cond / where(active) emulated via device 'err' scalar
// guarding all writes to carried state (u, v, err).
// ---------------------------------------------------------------------------

typedef __attribute__((ext_vector_type(16))) __bf16 v16bf;
typedef __attribute__((ext_vector_type(8)))  float  v8f;

static constexpr int NB  = 256;     // batch
static constexpr int NV  = 30000;   // vocab
static constexpr int NH  = 512;     // hidden
static constexpr int NK  = 200;     // topics
static constexpr int NE  = 300;     // embed dim
static constexpr int VP  = 30016;   // NV padded to mult of 32 (inner dim)
static constexpr int KP  = 224;     // NK padded to mult of 32 (inner dim)
static constexpr int MP  = 208;     // NK padded to mult of 16 (output rows)
static constexpr int NSTEPS_Q = VP / 32;   // 938
#define SH_THR 0.005f

union NstmFrag { v16bf v; uint4 q[2]; };

__device__ inline v16bf nstm_load_frag(const __bf16* rowp, int kk) {
  NstmFrag f;
  f.q[0] = *reinterpret_cast<const uint4*>(rowp + kk);
  f.q[1] = *reinterpret_cast<const uint4*>(rowp + kk + 16);
  return f.v;
}

// ---------------- GEMM-P: out[B,V] (f32) = A[B,KP](bf16) x Bt[V,KP](bf16)^T
// one 16x16 tile per wave, 4 waves/block, inner K = 224 (7 wmma steps)
__global__ void nstm_gemmP(const __bf16* __restrict__ A,
                           const __bf16* __restrict__ Bt,
                           float* __restrict__ out) {
  const int lane = threadIdx.x & 31;
  const int wave = threadIdx.x >> 5;
  const int tile = blockIdx.x * 4 + wave;      // 30000 tiles
  const int mtile = tile & 15;                 // 16 tiles over B
  const int ntile = tile >> 4;                 // 1875 tiles over V
  const int l15 = lane & 15;
  const int hi  = (lane >> 4) * 8;             // k sub-offset per half-wave

  const __bf16* arow = A  + (size_t)(mtile * 16 + l15) * KP + hi;
  const __bf16* brow = Bt + (size_t)(ntile * 16 + l15) * KP + hi;

  v8f acc = {};
#pragma unroll
  for (int kk = 0; kk < KP; kk += 32) {
    v16bf a = nstm_load_frag(arow, kk);
    v16bf b = nstm_load_frag(brow, kk);
    acc = __builtin_amdgcn_wmma_f32_16x16x32_bf16(false, a, false, b,
                                                  (short)0, acc, false, false);
  }
  const int row0 = mtile * 16 + ((lane >> 4) << 3);
  float* o = out + (size_t)row0 * NV + ntile * 16 + l15;
#pragma unroll
  for (int r = 0; r < 8; ++r) o[(size_t)r * NV] = acc[r];
}

// ---------------- GEMM-Q: out[K,B] = A[MP,VP](bf16) x Bt[B,VP](bf16)^T
// one tile per block, 8 waves split-K + LDS reduce.
// mode 0: out = aKB / acc (guarded by err if errp!=null). mode 1: out = acc.
__global__ void nstm_gemmQ(const __bf16* __restrict__ A,
                           const __bf16* __restrict__ Bt,
                           const float* __restrict__ aKB,
                           float* __restrict__ out,
                           const float* __restrict__ errp,
                           int mode) {
  __shared__ float red[8][256];
  const int lane = threadIdx.x & 31;
  const int wave = threadIdx.x >> 5;
  const int mtile = blockIdx.x % 13;           // 13 tiles over MP=208
  const int ntile = blockIdx.x / 13;           // 16 tiles over B=256
  const int l15 = lane & 15;
  const int hi  = (lane >> 4) * 8;

  const __bf16* arow = A  + (size_t)(mtile * 16 + l15) * VP + hi;
  const __bf16* brow = Bt + (size_t)(ntile * 16 + l15) * VP + hi;

  v8f acc = {};
  for (int s = wave; s < NSTEPS_Q; s += 8) {
    const int kk = s * 32;
    v16bf a = nstm_load_frag(arow, kk);
    v16bf b = nstm_load_frag(brow, kk);
    acc = __builtin_amdgcn_wmma_f32_16x16x32_bf16(false, a, false, b,
                                                  (short)0, acc, false, false);
  }
#pragma unroll
  for (int r = 0; r < 8; ++r) red[wave][lane * 8 + r] = acc[r];
  __syncthreads();

  const int t = threadIdx.x;                   // 0..255, owns one tile element
  float sum = 0.f;
#pragma unroll
  for (int w = 0; w < 8; ++w) sum += red[w][t];
  const int l = t >> 3, r = t & 7;
  const int row = mtile * 16 + r + ((l >= 16) ? 8 : 0);
  const int col = ntile * 16 + (l & 15);
  if (row < NK) {
    if (mode == 0) {
      const float val = aKB[row * NB + col] / sum;
      const bool active = (errp == nullptr) || (*errp > SH_THR);
      if (active) out[row * NB + col] = val;
    } else {
      out[row * NB + col] = sum;
    }
  }
}

// ---------------- small helpers -------------------------------------------
__global__ void nstm_u_to_uT(const float* __restrict__ u, __bf16* __restrict__ uT) {
  int i = blockIdx.x * 256 + threadIdx.x;      // NB*NK
  if (i >= NB * NK) return;
  int b = i / NK, k = i % NK;
  uT[b * KP + k] = (__bf16)u[k * NB + b];
}

// wT[b,v] = dw[b,v] / t[b,v]  (bf16, stride VP); optional guarded copy to out2
__global__ void nstm_ratio(const float* __restrict__ dw, const float* __restrict__ t,
                           __bf16* __restrict__ out, __bf16* __restrict__ out2,
                           const float* __restrict__ errp) {
  int i = blockIdx.x * 256 + threadIdx.x;      // NB*NV
  if (i >= NB * NV) return;
  int b = i / NV, v = i % NV;
  float val = dw[i] / t[i];
  __bf16 h = (__bf16)val;
  out[(size_t)b * VP + v] = h;
  if (out2 != nullptr && *errp > SH_THR) out2[(size_t)b * VP + v] = h;
}

// colsum[b] = sum_v | dw/t2 * t3 - dw |
__global__ void nstm_colsum(const float* __restrict__ dw, const float* __restrict__ t2,
                            const float* __restrict__ t3, float* __restrict__ colsum) {
  __shared__ float red[256];
  const int b = blockIdx.x;
  float s = 0.f;
  for (int v = threadIdx.x; v < NV; v += 256) {
    size_t i = (size_t)b * NV + v;
    float d = dw[i];
    s += fabsf(d / t2[i] * t3[i] - d);
  }
  red[threadIdx.x] = s; __syncthreads();
  for (int st = 128; st; st >>= 1) {
    if (threadIdx.x < st) red[threadIdx.x] += red[threadIdx.x + st];
    __syncthreads();
  }
  if (threadIdx.x == 0) colsum[b] = red[0];
}

__global__ void nstm_err_final(const float* __restrict__ colsum, float* __restrict__ err) {
  __shared__ float red[256];
  red[threadIdx.x] = colsum[threadIdx.x];
  __syncthreads();
  for (int st = 128; st; st >>= 1) {
    if (threadIdx.x < st) red[threadIdx.x] = fmaxf(red[threadIdx.x], red[threadIdx.x + st]);
    __syncthreads();
  }
  if (threadIdx.x == 0 && *err > SH_THR) *err = red[0];
}

__global__ void nstm_init_u(float* u) {
  int i = blockIdx.x * 256 + threadIdx.x;
  if (i < NK * NB) u[i] = 1.0f / (float)NK;
}
__global__ void nstm_init_vT(__bf16* vT) {
  int i = blockIdx.x * 256 + threadIdx.x;
  if (i >= NB * VP) return;
  vT[i] = ((i % VP) < NV) ? (__bf16)1.0f : (__bf16)0.0f;
}
__global__ void nstm_init_err(float* err) { *err = 1.0f; }

// ---------------- setup: embeddings -> kernel matrices --------------------
__global__ void nstm_inv_norms(const float* __restrict__ emb, float* __restrict__ inv, int n) {
  int i = blockIdx.x * 256 + threadIdx.x;
  if (i >= n) return;
  const float* r = emb + (size_t)i * NE;
  float s = 0.f;
  for (int e = 0; e < NE; ++e) s += r[e] * r[e];
  inv[i] = 1.0f / fmaxf(sqrtf(s), 1e-12f);
}

__global__ void nstm_build_mats(const float* __restrict__ wemb, const float* __restrict__ temb,
                                const float* __restrict__ winv, const float* __restrict__ tinv,
                                __bf16* __restrict__ kerM, __bf16* __restrict__ kcM,
                                __bf16* __restrict__ kerT, __bf16* __restrict__ twT) {
  int i = blockIdx.x * 256 + threadIdx.x;      // NK*NV
  if (i >= NK * NV) return;
  int k = i / NV, v = i % NV;
  const float* tr = temb + (size_t)k * NE;
  const float* wr = wemb + (size_t)v * NE;
  float d = 0.f;
  for (int e = 0; e < NE; ++e) d += tr[e] * wr[e];
  float tw  = d * tinv[k] * winv[v];
  float c   = 1.0f - tw;
  float ker = __expf(-20.0f * c);
  kerM[(size_t)k * VP + v] = (__bf16)ker;
  kcM [(size_t)k * VP + v] = (__bf16)(ker * c);
  kerT[(size_t)v * KP + k] = (__bf16)ker;
  twT [(size_t)v * KP + k] = (__bf16)tw;
}

// ---------------- encoder --------------------------------------------------
__global__ void nstm_enc1(const float* __restrict__ bows, const float* __restrict__ W1,
                          const float* __restrict__ b1, float* __restrict__ h) {
  int i = blockIdx.x * 256 + threadIdx.x;      // NB*NH
  if (i >= NB * NH) return;
  int b = i >> 9, j = i & (NH - 1);
  const float4* x = reinterpret_cast<const float4*>(bows + (size_t)b * NV);
  const float4* w = reinterpret_cast<const float4*>(W1 + (size_t)j * NV);
  float acc = 0.f;
  for (int v = 0; v < NV / 4; ++v) {
    float4 a = x[v], c = w[v];
    acc += a.x * c.x + a.y * c.y + a.z * c.z + a.w * c.w;
  }
  acc += b1[j];
  h[i] = acc > 0.f ? acc : 0.f;
}

__global__ void nstm_enc2(const float* __restrict__ h, const float* __restrict__ W2,
                          const float* __restrict__ b2, float* __restrict__ z) {
  int i = blockIdx.x * 256 + threadIdx.x;      // NB*NK
  if (i >= NB * NK) return;
  int b = i / NK, k = i % NK;
  const float* hr = h + (size_t)b * NH;
  const float* wr = W2 + (size_t)k * NH;
  float acc = 0.f;
  for (int t = 0; t < NH; ++t) acc += hr[t] * wr[t];
  z[i] = acc + b2[k];
}

__global__ void nstm_bn_stats(const float* __restrict__ z, float* __restrict__ mu,
                              float* __restrict__ ivar) {
  int k = threadIdx.x;
  if (k >= NK) return;
  float s = 0.f, s2 = 0.f;
  for (int b = 0; b < NB; ++b) {
    float x = z[b * NK + k];
    s += x; s2 += x * x;
  }
  float m = s / (float)NB;
  float v = s2 / (float)NB - m * m;     // biased variance
  mu[k] = m;
  ivar[k] = rsqrtf(v + 1e-3f);
}

__global__ void nstm_softmax_z(const float* __restrict__ z, const float* __restrict__ mu,
                               const float* __restrict__ ivar, const float* __restrict__ gamma,
                               const float* __restrict__ beta, float* __restrict__ doc_topic,
                               float* __restrict__ aKB, __bf16* __restrict__ dtT) {
  __shared__ float red[256];
  __shared__ float zn[256];
  const int b = blockIdx.x, k = threadIdx.x;
  float x = -1e30f;
  if (k < NK) {
    x = gamma[k] * (z[b * NK + k] - mu[k]) * ivar[k] + beta[k];
    zn[k] = x;
  }
  red[k] = x; __syncthreads();
  for (int st = 128; st; st >>= 1) {
    if (k < st) red[k] = fmaxf(red[k], red[k + st]);
    __syncthreads();
  }
  float m = red[0]; __syncthreads();
  float e = (k < NK) ? __expf(zn[k] - m) : 0.f;
  red[k] = e; __syncthreads();
  for (int st = 128; st; st >>= 1) {
    if (k < st) red[k] += red[k + st];
    __syncthreads();
  }
  float S = red[0];
  if (k < NK) {
    float p = e / S;
    doc_topic[b * NK + k] = p;
    aKB[k * NB + b] = p;
    dtT[b * KP + k] = (__bf16)p;
  }
}

// row softmax over V: dw = softmax(bows, axis=1)
__global__ void nstm_softmax_rows(const float* __restrict__ x, float* __restrict__ out) {
  __shared__ float red[256];
  const int b = blockIdx.x;
  const float* xr = x + (size_t)b * NV;
  float m = -1e30f;
  for (int v = threadIdx.x; v < NV; v += 256) m = fmaxf(m, xr[v]);
  red[threadIdx.x] = m; __syncthreads();
  for (int st = 128; st; st >>= 1) {
    if (threadIdx.x < st) red[threadIdx.x] = fmaxf(red[threadIdx.x], red[threadIdx.x + st]);
    __syncthreads();
  }
  m = red[0]; __syncthreads();
  float s = 0.f;
  for (int v = threadIdx.x; v < NV; v += 256) s += __expf(xr[v] - m);
  red[threadIdx.x] = s; __syncthreads();
  for (int st = 128; st; st >>= 1) {
    if (threadIdx.x < st) red[threadIdx.x] += red[threadIdx.x + st];
    __syncthreads();
  }
  float inv = 1.0f / red[0];
  for (int v = threadIdx.x; v < NV; v += 256)
    out[(size_t)b * NV + v] = __expf(xr[v] - m) * inv;
}

// per-row: recvec[b] = dot(rec,bows) - logsumexp(rec) * sum(bows)
__global__ void nstm_rec_rows(const float* __restrict__ rec, const float* __restrict__ bows,
                              float* __restrict__ recvec) {
  __shared__ float red[256];
  const int b = blockIdx.x;
  const float* rr = rec + (size_t)b * NV;
  const float* br = bows + (size_t)b * NV;
  float m = -1e30f;
  for (int v = threadIdx.x; v < NV; v += 256) m = fmaxf(m, rr[v]);
  red[threadIdx.x] = m; __syncthreads();
  for (int st = 128; st; st >>= 1) {
    if (threadIdx.x < st) red[threadIdx.x] = fmaxf(red[threadIdx.x], red[threadIdx.x + st]);
    __syncthreads();
  }
  m = red[0]; __syncthreads();
  float s = 0.f, dot = 0.f, ws = 0.f;
  for (int v = threadIdx.x; v < NV; v += 256) {
    float x = rr[v], w = br[v];
    s += __expf(x - m);
    dot += x * w;
    ws  += w;
  }
  red[threadIdx.x] = s; __syncthreads();
  for (int st = 128; st; st >>= 1) { if (threadIdx.x < st) red[threadIdx.x] += red[threadIdx.x + st]; __syncthreads(); }
  float S = red[0]; __syncthreads();
  red[threadIdx.x] = dot; __syncthreads();
  for (int st = 128; st; st >>= 1) { if (threadIdx.x < st) red[threadIdx.x] += red[threadIdx.x + st]; __syncthreads(); }
  float D = red[0]; __syncthreads();
  red[threadIdx.x] = ws; __syncthreads();
  for (int st = 128; st; st >>= 1) { if (threadIdx.x < st) red[threadIdx.x] += red[threadIdx.x + st]; __syncthreads(); }
  float W = red[0];
  if (threadIdx.x == 0) recvec[b] = D - (m + logf(S)) * W;
}

__global__ void nstm_final_rec(const float* __restrict__ recvec, float* __restrict__ out) {
  __shared__ float red[256];
  red[threadIdx.x] = recvec[threadIdx.x];
  __syncthreads();
  for (int st = 128; st; st >>= 1) { if (threadIdx.x < st) red[threadIdx.x] += red[threadIdx.x + st]; __syncthreads(); }
  if (threadIdx.x == 0) out[0] = 0.7f * (-(red[0] / (float)NB));
}

__global__ void nstm_final_sink(const float* __restrict__ u, const float* __restrict__ q,
                                float* __restrict__ out) {
  __shared__ float red[256];
  const int b = threadIdx.x;
  float s = 0.f;
  for (int k = 0; k < NK; ++k) s += u[k * NB + b] * q[k * NB + b];
  red[b] = s; __syncthreads();
  for (int st = 128; st; st >>= 1) { if (threadIdx.x < st) red[threadIdx.x] += red[threadIdx.x + st]; __syncthreads(); }
  if (threadIdx.x == 0) out[1] = red[0] / (float)NB;
}

// ---------------------------------------------------------------------------
extern "C" void kernel_launch(void* const* d_in, const int* in_sizes, int n_in,
                              void* d_out, int out_size, void* d_ws, size_t ws_size,
                              hipStream_t stream) {
  (void)in_sizes; (void)n_in; (void)out_size; (void)ws_size;
  const float* bows  = (const float*)d_in[0];
  const float* W1    = (const float*)d_in[1];
  const float* b1    = (const float*)d_in[2];
  const float* W2    = (const float*)d_in[3];
  const float* b2    = (const float*)d_in[4];
  const float* gamma = (const float*)d_in[5];
  const float* beta  = (const float*)d_in[6];
  const float* wemb  = (const float*)d_in[7];
  const float* temb  = (const float*)d_in[8];
  float* out = (float*)d_out;

  char* p = (char*)d_ws;
  auto alloc = [&](size_t bytes) -> char* {
    char* r = p;
    p += (bytes + 255) & ~(size_t)255;
    return r;
  };

  // f32 buffers
  float* tA   = (float*)alloc((size_t)NB * NV * 4);   // [B,V] temp / rec logits / t1 / t3
  float* tB   = (float*)alloc((size_t)NB * NV * 4);   // [B,V] temp t2
  float* dw   = (float*)alloc((size_t)NB * NV * 4);   // doc_word softmax
  float* u    = (float*)alloc((size_t)NK * NB * 4);
  float* u1b  = (float*)alloc((size_t)NK * NB * 4);
  float* aKB  = (float*)alloc((size_t)NK * NB * 4);
  float* q    = (float*)alloc((size_t)NK * NB * 4);
  float* h    = (float*)alloc((size_t)NB * NH * 4);
  float* z    = (float*)alloc((size_t)NB * NK * 4);
  float* dt   = (float*)alloc((size_t)NB * NK * 4);
  float* mu   = (float*)alloc(NK * 4);
  float* ivar = (float*)alloc(NK * 4);
  float* winv = (float*)alloc(NV * 4);
  float* tinv = (float*)alloc(NK * 4);
  float* colsum = (float*)alloc(NB * 4);
  float* recvec = (float*)alloc(NB * 4);
  float* err    = (float*)alloc(256);
  // bf16 buffers
  __bf16* kerM = (__bf16*)alloc((size_t)MP * VP * 2);  // kernel [MP,VP]
  __bf16* kcM  = (__bf16*)alloc((size_t)MP * VP * 2);  // kernel*cost
  __bf16* kerT = (__bf16*)alloc((size_t)NV * KP * 2);  // kernel^T [V,KP]
  __bf16* twT  = (__bf16*)alloc((size_t)NV * KP * 2);  // topic_word^T
  __bf16* wT   = (__bf16*)alloc((size_t)NB * VP * 2);  // (b/t)^T scratch
  __bf16* v1T  = (__bf16*)alloc((size_t)NB * VP * 2);  // v1^T scratch
  __bf16* vT   = (__bf16*)alloc((size_t)NB * VP * 2);  // carried v^T
  __bf16* uT   = (__bf16*)alloc((size_t)NB * KP * 2);
  __bf16* dtT  = (__bf16*)alloc((size_t)NB * KP * 2);

  // ---- zero-fill padded bf16 arrays (valid regions overwritten below) ----
  hipMemsetAsync(kerM, 0, (size_t)MP * VP * 2, stream);
  hipMemsetAsync(kcM,  0, (size_t)MP * VP * 2, stream);
  hipMemsetAsync(kerT, 0, (size_t)NV * KP * 2, stream);
  hipMemsetAsync(twT,  0, (size_t)NV * KP * 2, stream);
  hipMemsetAsync(wT,   0, (size_t)NB * VP * 2, stream);
  hipMemsetAsync(v1T,  0, (size_t)NB * VP * 2, stream);
  hipMemsetAsync(uT,   0, (size_t)NB * KP * 2, stream);
  hipMemsetAsync(dtT,  0, (size_t)NB * KP * 2, stream);

  nstm_init_err<<<1, 1, 0, stream>>>(err);
  nstm_init_u<<<(NK * NB + 255) / 256, 256, 0, stream>>>(u);
  nstm_init_vT<<<(NB * VP + 255) / 256, 256, 0, stream>>>(vT);

  // ---- embeddings -> cost/kernel matrices ----
  nstm_inv_norms<<<(NV + 255) / 256, 256, 0, stream>>>(wemb, winv, NV);
  nstm_inv_norms<<<1, 256, 0, stream>>>(temb, tinv, NK);
  nstm_build_mats<<<((size_t)NK * NV + 255) / 256, 256, 0, stream>>>(
      wemb, temb, winv, tinv, kerM, kcM, kerT, twT);

  // ---- encoder -> doc_topic (and a = doc_topic^T) ----
  nstm_enc1<<<(NB * NH + 255) / 256, 256, 0, stream>>>(bows, W1, b1, h);
  nstm_enc2<<<(NB * NK + 255) / 256, 256, 0, stream>>>(h, W2, b2, z);
  nstm_bn_stats<<<1, 256, 0, stream>>>(z, mu, ivar);
  nstm_softmax_z<<<NB, 256, 0, stream>>>(z, mu, ivar, gamma, beta, dt, aKB, dtT);

  // ---- doc_word = softmax(bows) ----
  nstm_softmax_rows<<<NB, 256, 0, stream>>>(bows, dw);

  // ---- reconstruction loss: rec = doc_topic @ topic_word, log_softmax ----
  nstm_gemmP<<<(16 * (NV / 16)) / 4, 128, 0, stream>>>(dtT, twT, tA);
  nstm_rec_rows<<<NB, 256, 0, stream>>>(tA, bows, recvec);
  nstm_final_rec<<<1, 256, 0, stream>>>(recvec, out);

  // ---- Sinkhorn loop (100 fixed iterations, checks at i%20==0) ----
  const int gP = (16 * (NV / 16)) / 4;   // 7500 blocks
  const int gQ = 13 * 16;                // 208 blocks
  const int gE = (NB * NV + 255) / 256;  // elementwise over [B,V]
  for (int i = 0; i < 100; ++i) {
    if (i % 20 == 0) {
      // t1 = u^T K ; w = b/t1 ; u1 = a/(K w)   (u1 is a temp -> unguarded)
      nstm_u_to_uT<<<(NB * NK + 255) / 256, 256, 0, stream>>>(u, uT);
      nstm_gemmP<<<gP, 128, 0, stream>>>(uT, kerT, tA);
      nstm_ratio<<<gE, 256, 0, stream>>>(dw, tA, wT, nullptr, nullptr);
      nstm_gemmQ<<<gQ, 256, 0, stream>>>(kerM, wT, aKB, u1b, nullptr, 0);
      // t2 = K^T u1 ; v1 = b/t2 (guarded copy into carried v)
      nstm_u_to_uT<<<(NB * NK + 255) / 256, 256, 0, stream>>>(u1b, uT);
      nstm_gemmP<<<gP, 128, 0, stream>>>(uT, kerT, tB);
      nstm_ratio<<<gE, 256, 0, stream>>>(dw, tB, v1T, vT, err);
      // u2 = a/(K v1)  (guarded write into carried u)
      nstm_gemmQ<<<gQ, 256, 0, stream>>>(kerM, v1T, aKB, u, err, 0);
      // t3 = K^T u ; err_new = max_b sum_v |v1*t3 - b|  (guarded err update)
      nstm_u_to_uT<<<(NB * NK + 255) / 256, 256, 0, stream>>>(u, uT);
      nstm_gemmP<<<gP, 128, 0, stream>>>(uT, kerT, tA);
      nstm_colsum<<<NB, 256, 0, stream>>>(dw, tB, tA, colsum);
      nstm_err_final<<<1, 256, 0, stream>>>(colsum, err);
    } else {
      nstm_u_to_uT<<<(NB * NK + 255) / 256, 256, 0, stream>>>(u, uT);
      nstm_gemmP<<<gP, 128, 0, stream>>>(uT, kerT, tA);
      nstm_ratio<<<gE, 256, 0, stream>>>(dw, tA, wT, nullptr, nullptr);
      nstm_gemmQ<<<gQ, 256, 0, stream>>>(kerM, wT, aKB, u, err, 0);
    }
  }

  // ---- sinkhorn loss: mean_b sum_k u[k,b] * ((kernel*cost) @ v)[k,b] ----
  nstm_gemmQ<<<gQ, 256, 0, stream>>>(kcM, vT, nullptr, q, nullptr, 1);
  nstm_final_sink<<<1, 256, 0, stream>>>(u, q, out);
}
